// SiglipAttention_57131654972075
// MI455X (gfx1250) — compile-verified, matrix operand order checked by reading, not run on hardware
//
#include <hip/hip_runtime.h>
#include <cstddef>

typedef __attribute__((ext_vector_type(16))) _Float16 v16h;
typedef __attribute__((ext_vector_type(8)))  _Float16 v8h;
typedef __attribute__((ext_vector_type(8)))  float    v8f;

namespace cfg {
constexpr int B = 16, S = 729, E = 1152, H = 16, D = 72;
constexpr int BS = B * S;                          // 11664 (multiple of 16)
constexpr size_t BSE  = (size_t)BS * E;            // 13,436,928
constexpr size_t EE   = (size_t)E * E;             // 1,327,104
constexpr int SPK = 768;   // Q/K padded seq rows (multiple of 64-wide N strip)
constexpr int SPV = 736;   // P cols / Pq rows / Vt cols pad (46*16, multiple of 32)
constexpr int DP  = 96;    // padded head dim (multiple of 32)
constexpr float SCALE = 0.11785113019775792f;      // 72^-0.5
}

#define WMMA_F16(a, b, c) \
  __builtin_amdgcn_wmma_f32_16x16x32_f16(false, (a), false, (b), (short)0, (c), false, false)

// ---------------------------------------------------------------------------
// Utility kernels
// ---------------------------------------------------------------------------

__global__ void init_slots_kernel(float* slots) {
  if (threadIdx.x < 16) slots[threadIdx.x] = 0.0f;
}

__global__ void absmax_kernel(const float* __restrict__ x, size_t n,
                              float* __restrict__ slot) {
  __shared__ float red[256];
  float m = 0.0f;
  for (size_t i = (size_t)blockIdx.x * blockDim.x + threadIdx.x; i < n;
       i += (size_t)gridDim.x * blockDim.x)
    m = fmaxf(m, fabsf(x[i]));
  red[threadIdx.x] = m;
  __syncthreads();
  for (int s = 128; s > 0; s >>= 1) {
    if ((int)threadIdx.x < s)
      red[threadIdx.x] = fmaxf(red[threadIdx.x], red[threadIdx.x + s]);
    __syncthreads();
  }
  if (threadIdx.x == 0)
    atomicMax((unsigned int*)slot, __float_as_uint(red[0]));
}

// fq(x,bits) forward value -> fp16, flat layout
__global__ void quant_f32_to_f16_kernel(const float* __restrict__ x,
                                        _Float16* __restrict__ y, size_t n,
                                        const float* __restrict__ amax, float qmax) {
  float s = fmaxf(amax[0], 1e-8f) / qmax;
  float inv = 1.0f / s;
  for (size_t i = (size_t)blockIdx.x * blockDim.x + threadIdx.x; i < n;
       i += (size_t)gridDim.x * blockDim.x) {
    float r = rintf(x[i] * inv);
    r = fminf(fmaxf(r, -qmax), qmax);
    y[i] = (_Float16)(r * s);
  }
}

// fq(q/k,8): (B,S,E) fp32 -> per-head padded (B,H,SPK,DP) fp16, zero pads
__global__ void quant_pack_qk_kernel(const float* __restrict__ X,
                                     _Float16* __restrict__ Y,
                                     const float* __restrict__ amax, float qmax) {
  using namespace cfg;
  const size_t total = (size_t)B * H * SPK * DP;
  float s = fmaxf(amax[0], 1e-8f) / qmax;
  float inv = 1.0f / s;
  for (size_t i = (size_t)blockIdx.x * blockDim.x + threadIdx.x; i < total;
       i += (size_t)gridDim.x * blockDim.x) {
    int dp = (int)(i % DP);
    size_t t = i / DP;
    int sp = (int)(t % SPK);
    t /= SPK;
    int h = (int)(t % H);
    int b = (int)(t / H);
    float v = 0.0f;
    if (sp < S && dp < D) {
      float x = X[((size_t)b * S + sp) * E + (size_t)h * D + dp];
      float r = rintf(x * inv);
      r = fminf(fmaxf(r, -qmax), qmax);
      v = r * s;
    }
    Y[i] = (_Float16)v;
  }
}

// fq(v,8): (B,S,E) fp32 -> per-head TRANSPOSED padded (B,H,DP,SPV) fp16
__global__ void quant_pack_vt_kernel(const float* __restrict__ X,
                                     _Float16* __restrict__ Y,
                                     const float* __restrict__ amax, float qmax) {
  using namespace cfg;
  const size_t total = (size_t)B * H * DP * SPV;
  float s = fmaxf(amax[0], 1e-8f) / qmax;
  float inv = 1.0f / s;
  for (size_t i = (size_t)blockIdx.x * blockDim.x + threadIdx.x; i < total;
       i += (size_t)gridDim.x * blockDim.x) {
    int sp = (int)(i % SPV);
    size_t t = i / SPV;
    int dp = (int)(t % DP);
    t /= DP;
    int h = (int)(t % H);
    int b = (int)(t / H);
    float v = 0.0f;
    if (sp < S && dp < D) {
      float x = X[((size_t)b * S + sp) * E + (size_t)h * D + dp];
      float r = rintf(x * inv);
      r = fminf(fmaxf(r, -qmax), qmax);
      v = r * s;
    }
    Y[i] = (_Float16)v;
  }
}

// Row softmax over S entries; fused global max(probs) = max_rows(1/sum)
__global__ void softmax_rows_kernel(float* __restrict__ P,
                                    float* __restrict__ amax_slot) {
  using namespace cfg;
  float* p = P + (size_t)blockIdx.x * S;
  __shared__ float red[256];
  int tid = threadIdx.x;
  float m = -3.4e38f;
  for (int i = tid; i < S; i += 256) m = fmaxf(m, p[i]);
  red[tid] = m;
  __syncthreads();
  for (int s = 128; s > 0; s >>= 1) {
    if (tid < s) red[tid] = fmaxf(red[tid], red[tid + s]);
    __syncthreads();
  }
  m = red[0];
  __syncthreads();
  float sum = 0.0f;
  for (int i = tid; i < S; i += 256) {
    float e = __expf(p[i] - m);
    p[i] = e;
    sum += e;
  }
  red[tid] = sum;
  __syncthreads();
  for (int s = 128; s > 0; s >>= 1) {
    if (tid < s) red[tid] += red[tid + s];
    __syncthreads();
  }
  float inv = 1.0f / red[0];
  for (int i = tid; i < S; i += 256) p[i] *= inv;
  if (tid == 0) atomicMax((unsigned int*)amax_slot, __float_as_uint(inv));
}

// fq(probs,16): update fp32 probs in place (returned attn_weights) AND write a
// padded fp16 copy (B*H, SPV, SPV) for the PV WMMA. One block per padded row.
__global__ void quant_pack_p_kernel(float* __restrict__ P, _Float16* __restrict__ Pq,
                                    const float* __restrict__ amax, float qmax) {
  using namespace cfg;
  int rowp = blockIdx.x;            // bh*SPV + qp
  int qp = rowp % SPV;
  int bh = rowp / SPV;
  float s = fmaxf(amax[0], 1e-8f) / qmax;
  float inv = 1.0f / s;
  _Float16* yq = Pq + (size_t)rowp * SPV;
  if (qp < S) {
    float* pr = P + ((size_t)bh * S + qp) * S;
    for (int k = threadIdx.x; k < SPV; k += 256) {
      float v = 0.0f;
      if (k < S) {
        float r = rintf(pr[k] * inv);
        r = fminf(fmaxf(r, -qmax), qmax);
        v = r * s;
        pr[k] = v;
      }
      yq[k] = (_Float16)v;
    }
  } else {
    for (int k = threadIdx.x; k < SPV; k += 256) yq[k] = (_Float16)0.0f;
  }
}

// ---------------------------------------------------------------------------
// WMMA fragment load: unconditional, 16B-aligned, two b128 loads per lane.
// Lane L: row = L%16; K halves {k0..k0+7, k0+16..k0+23}, k0 = 8*(L>=16).
// Same mapping serves A (row-major MxK) and B (W row n = B column n).
// ---------------------------------------------------------------------------
__device__ __forceinline__ v16h frag_ld(const _Float16* __restrict__ base,
                                        int lda, int r0, int k0) {
  int lane = threadIdx.x & 31;
  const _Float16* p =
      base + (size_t)(r0 + (lane & 15)) * lda + k0 + (lane >> 4) * 8;
  union { v16h v; _Float16 s[16]; } u;
  *(v8h*)(u.s)     = *(const v8h*)(p);
  *(v8h*)(u.s + 8) = *(const v8h*)(p + 16);
  return u.v;
}

// ---------------------------------------------------------------------------
// GEMM kernels — one wave per block, no bounds checks in the k-loop
// ---------------------------------------------------------------------------

// Y[BS,E] = X[BS,E] * W[E,E]^T + bias; 16x64 per wave; optional fused abs-max.
__global__ void gemm_xwt_bias_kernel(const _Float16* __restrict__ X,
                                     const _Float16* __restrict__ W,
                                     const float* __restrict__ bias,
                                     float* __restrict__ Y,
                                     float* __restrict__ amax_slot) {
  using namespace cfg;
  int lane = threadIdx.x & 31;
  int m0 = blockIdx.x * 16;
  int n0 = blockIdx.y * 64;
  v8f acc[4] = {};
  for (int kt = 0; kt < E; kt += 32) {
    v16h a = frag_ld(X, E, m0, kt);
#pragma unroll
    for (int t = 0; t < 4; ++t) {
      v16h b = frag_ld(W, E, n0 + t * 16, kt);
      acc[t] = WMMA_F16(a, b, acc[t]);
    }
  }
  int rb = (lane >> 4) * 8;
  float lmax = 0.0f;
#pragma unroll
  for (int t = 0; t < 4; ++t) {
    union { v8f v; float f[8]; } ua;
    ua.v = acc[t];
    int col = n0 + t * 16 + (lane & 15);
    float bval = bias[col];
#pragma unroll
    for (int r = 0; r < 8; ++r) {
      int row = m0 + rb + r;            // always < BS
      float v = ua.f[r] + bval;
      Y[(size_t)row * E + col] = v;
      lmax = fmaxf(lmax, fabsf(v));
    }
  }
  if (amax_slot) atomicMax((unsigned int*)amax_slot, __float_as_uint(lmax));
}

// P[bh,q,k] = SCALE * sum_d Qp[bh,q,d]*Kp[bh,k,d]; padded (SPK,DP) operands.
__global__ void qk_kernel(const _Float16* __restrict__ Qp,
                          const _Float16* __restrict__ Kp,
                          float* __restrict__ P, float scale) {
  using namespace cfg;
  int lane = threadIdx.x & 31;
  int bh = blockIdx.z;
  int q0 = blockIdx.x * 16;
  int n0 = blockIdx.y * 64;             // key position strip (0..704)
  const _Float16* Ab = Qp + (size_t)bh * SPK * DP;
  const _Float16* Bb = Kp + (size_t)bh * SPK * DP;
  v8f acc[4] = {};
  for (int kt = 0; kt < DP; kt += 32) { // 3 steps, D=72 zero-padded to 96
    v16h a = frag_ld(Ab, DP, q0, kt);
#pragma unroll
    for (int t = 0; t < 4; ++t) {
      v16h b = frag_ld(Bb, DP, n0 + t * 16, kt);
      acc[t] = WMMA_F16(a, b, acc[t]);
    }
  }
  float* Pr = P + (size_t)bh * S * S;
  int rb = (lane >> 4) * 8;
#pragma unroll
  for (int t = 0; t < 4; ++t) {
    union { v8f v; float f[8]; } ua;
    ua.v = acc[t];
    int col = n0 + t * 16 + (lane & 15);
    if (col < S) {
#pragma unroll
      for (int r = 0; r < 8; ++r) {
        int row = q0 + rb + r;
        if (row < S) Pr[(size_t)row * S + col] = ua.f[r] * scale;
      }
    }
  }
}

// O[b,q,h*D+d] = sum_k Pq[bh,q,k]*Vt[bh,d,k]; 16x80 per wave; fused abs-max.
__global__ void pv_kernel(const _Float16* __restrict__ Pq,
                          const _Float16* __restrict__ Vt,
                          float* __restrict__ O,
                          float* __restrict__ amax_slot) {
  using namespace cfg;
  int lane = threadIdx.x & 31;
  int bh = blockIdx.z, b = bh >> 4, h = bh & 15;
  int q0 = blockIdx.x * 16;
  const _Float16* Ab = Pq + (size_t)bh * SPV * SPV;
  const _Float16* Bb = Vt + (size_t)bh * DP * SPV;
  v8f acc[5] = {};
  for (int kt = 0; kt < SPV; kt += 32) { // 23 steps, S=729 zero-padded to 736
    v16h a = frag_ld(Ab, SPV, q0, kt);
#pragma unroll
    for (int t = 0; t < 5; ++t) {        // d rows 0..79 (< DP=96, pad zeroed)
      v16h bfr = frag_ld(Bb, SPV, t * 16, kt);
      acc[t] = WMMA_F16(a, bfr, acc[t]);
    }
  }
  int rb = (lane >> 4) * 8;
  float lmax = 0.0f;
#pragma unroll
  for (int t = 0; t < 5; ++t) {
    union { v8f v; float f[8]; } ua;
    ua.v = acc[t];
    int col = t * 16 + (lane & 15);
    if (col < D) {
#pragma unroll
      for (int r = 0; r < 8; ++r) {
        int row = q0 + rb + r;
        if (row < S) {
          float v = ua.f[r];
          O[((size_t)b * S + row) * E + (size_t)h * D + col] = v;
          lmax = fmaxf(lmax, fabsf(v));
        }
      }
    }
  }
  atomicMax((unsigned int*)amax_slot, __float_as_uint(lmax));
}

// ---------------------------------------------------------------------------
// Host launch
// ---------------------------------------------------------------------------

extern "C" void kernel_launch(void* const* d_in, const int* in_sizes, int n_in,
                              void* d_out, int out_size, void* d_ws, size_t ws_size,
                              hipStream_t stream) {
  using namespace cfg;
  (void)in_sizes; (void)n_in; (void)out_size; (void)ws_size;

  const float* hidden = (const float*)d_in[0];
  const float* wq = (const float*)d_in[1];
  const float* bq = (const float*)d_in[2];
  const float* wk = (const float*)d_in[3];
  const float* bk = (const float*)d_in[4];
  const float* wv = (const float*)d_in[5];
  const float* bvv = (const float*)d_in[6];
  const float* wo = (const float*)d_in[7];
  const float* bo = (const float*)d_in[8];

  float* out_attn  = (float*)d_out;      // (B,S,E)
  float* out_probs = out_attn + BSE;     // (B,H,S,S)

  char* w = (char*)d_ws;
  size_t off = 0;
  auto take = [&](size_t bytes) -> char* {
    char* p = w + off;
    off = (off + bytes + 255) & ~(size_t)255;
    return p;
  };
  const size_t QK_PACK = (size_t)B * H * SPK * DP;  // 18,874,368 halves
  const size_t VT_PACK = (size_t)B * H * DP * SPV;  // 18,087,936 halves
  const size_t PQ_PACK = (size_t)B * H * SPV * SPV; // 138,674,176 halves

  float*    slots = (float*)take(16 * sizeof(float));
  _Float16* Xh  = (_Float16*)take(BSE * 2);
  _Float16* Wqh = (_Float16*)take(EE * 2);
  _Float16* Wkh = (_Float16*)take(EE * 2);
  _Float16* Wvh = (_Float16*)take(EE * 2);
  _Float16* Woh = (_Float16*)take(EE * 2);
  float* Qf = (float*)take(BSE * 4);
  float* Kf = (float*)take(BSE * 4);
  float* Vf = (float*)take(BSE * 4);
  _Float16* Qqp = (_Float16*)take(QK_PACK * 2);
  _Float16* Kqp = (_Float16*)take(QK_PACK * 2);
  _Float16* Vt  = (_Float16*)take(VT_PACK * 2);
  _Float16* Pq  = (_Float16*)take(PQ_PACK * 2);
  float*    AttnF = Qf;                  // fp32 Q dead after pack
  _Float16* AttnH = (_Float16*)Kf;       // fp32 K dead after pack

  dim3 blk256(256), blk32(32);
  auto gsz = [](size_t n) {
    size_t g = (n + 255) / 256;
    return (unsigned)(g > 4096 ? 4096 : g);
  };

  // scales: 0=hidden 1..4=wq,wk,wv,wo 5..7=q,k,v 8=probs 9=attn
  init_slots_kernel<<<1, 32, 0, stream>>>(slots);

  absmax_kernel<<<gsz(BSE), blk256, 0, stream>>>(hidden, BSE, slots + 0);
  absmax_kernel<<<gsz(EE),  blk256, 0, stream>>>(wq, EE, slots + 1);
  absmax_kernel<<<gsz(EE),  blk256, 0, stream>>>(wk, EE, slots + 2);
  absmax_kernel<<<gsz(EE),  blk256, 0, stream>>>(wv, EE, slots + 3);
  absmax_kernel<<<gsz(EE),  blk256, 0, stream>>>(wo, EE, slots + 4);

  quant_f32_to_f16_kernel<<<gsz(BSE), blk256, 0, stream>>>(hidden, Xh, BSE, slots + 0, 32767.f);
  quant_f32_to_f16_kernel<<<gsz(EE),  blk256, 0, stream>>>(wq, Wqh, EE, slots + 1, 32767.f);
  quant_f32_to_f16_kernel<<<gsz(EE),  blk256, 0, stream>>>(wk, Wkh, EE, slots + 2, 32767.f);
  quant_f32_to_f16_kernel<<<gsz(EE),  blk256, 0, stream>>>(wv, Wvh, EE, slots + 3, 32767.f);
  quant_f32_to_f16_kernel<<<gsz(EE),  blk256, 0, stream>>>(wo, Woh, EE, slots + 4, 32767.f);

  // q/k/v projections (no bounds anywhere: 11664x1152x1152), fused abs-max
  dim3 gproj(BS / 16, E / 64);           // 729 x 18, 1 wave/block
  gemm_xwt_bias_kernel<<<gproj, blk32, 0, stream>>>(Xh, Wqh, bq, Qf, slots + 5);
  gemm_xwt_bias_kernel<<<gproj, blk32, 0, stream>>>(Xh, Wkh, bk, Kf, slots + 6);
  gemm_xwt_bias_kernel<<<gproj, blk32, 0, stream>>>(Xh, Wvh, bvv, Vf, slots + 7);

  // fq(*,8) + pack into padded per-head layouts
  quant_pack_qk_kernel<<<gsz(QK_PACK), blk256, 0, stream>>>(Qf, Qqp, slots + 5, 127.f);
  quant_pack_qk_kernel<<<gsz(QK_PACK), blk256, 0, stream>>>(Kf, Kqp, slots + 6, 127.f);
  quant_pack_vt_kernel<<<gsz(VT_PACK), blk256, 0, stream>>>(Vf, Vt, slots + 7, 127.f);

  // QK^T * SCALE -> logits into the attn_weights output region
  dim3 gqk(SPV / 16, SPK / 64, B * H);   // 46 x 12 x 256, 1 wave/block
  qk_kernel<<<gqk, blk32, 0, stream>>>(Qqp, Kqp, out_probs, SCALE);

  // softmax rows + fused global prob max (= 1/sum per row)
  softmax_rows_kernel<<<B * H * S, blk256, 0, stream>>>(out_probs, slots + 8);

  // fq(probs,16): finalize attn_weights in place + padded fp16 copy for PV
  quant_pack_p_kernel<<<B * H * SPV, blk256, 0, stream>>>(out_probs, Pq, slots + 8, 32767.f);

  // P x V -> attn output (B,S,E), fused abs-max
  dim3 gpv(SPV / 16, 1, B * H);          // 46 x 1 x 256, 1 wave/block
  pv_kernel<<<gpv, blk32, 0, stream>>>(Pq, Vt, AttnF, slots + 9);

  // fq(attn,16) -> fp16, then out projection + bias into d_out
  quant_f32_to_f16_kernel<<<gsz(BSE), blk256, 0, stream>>>(AttnF, AttnH, BSE, slots + 9, 32767.f);
  gemm_xwt_bias_kernel<<<gproj, blk32, 0, stream>>>(AttnH, Woh, bo, out_attn, nullptr);
}
